// MTA_sharemoe2_58634893525398
// MI455X (gfx1250) — matrix operand order repaired and checked by reading, not compiled, not run on hardware
//
#include <hip/hip_runtime.h>
#include <hip/hip_bf16.h>
#include <math.h>

#define B_ 8
#define S_ 2048
#define D_ 1024
#define F_ 4096
#define E_ 5

typedef __attribute__((ext_vector_type(16))) __bf16 v16bf;
typedef __attribute__((ext_vector_type(8)))  float  v8f;
typedef __attribute__((ext_vector_type(4)))  unsigned int v4u;
typedef __attribute__((ext_vector_type(8)))  int v8i;
typedef __attribute__((ext_vector_type(4)))  int v4i;

#if defined(__gfx1250__) && __has_builtin(__builtin_amdgcn_tensor_load_to_lds) && __has_builtin(__builtin_amdgcn_s_wait_tensorcnt)
#define USE_TDM 1
#endif

__device__ __forceinline__ unsigned short f32_to_bf16_rne(float f) {
  union { float f; unsigned u; } x; x.f = f;
  unsigned u = x.u;
  u += 0x7FFFu + ((u >> 16) & 1u);   // round-to-nearest-even
  return (unsigned short)(u >> 16);
}

// ---------------- f32 -> bf16 elementwise conversion ----------------
__global__ __launch_bounds__(256) void cvt_f32_bf16(const float* __restrict__ in,
                                                    unsigned short* __restrict__ out,
                                                    long n) {
  long stride = (long)gridDim.x * blockDim.x;
  for (long i = (long)blockIdx.x * blockDim.x + threadIdx.x; i < n; i += stride)
    out[i] = f32_to_bf16_rne(in[i]);
}

// ---------------- f32 (KxN) -> bf16 transposed (NxK), per matrix ----------------
__global__ __launch_bounds__(256) void cvt_transpose_f32_bf16(
    const float* __restrict__ in,      // nmat x (K x N) row-major
    unsigned short* __restrict__ out,  // nmat x (N x K) row-major
    int K, int N)
{
  __shared__ float tile[32][33];
  const int m  = blockIdx.z;
  const int k0 = blockIdx.x * 32;
  const int n0 = blockIdx.y * 32;
  const int tx = threadIdx.x & 31;
  const int ty = threadIdx.x >> 5;   // 0..7
  const float* I = in + (long)m * K * N;
  unsigned short* O = out + (long)m * K * N;
  #pragma unroll
  for (int j = 0; j < 4; ++j) {
    int k = ty * 4 + j;
    tile[k][tx] = I[(long)(k0 + k) * N + n0 + tx];
  }
  __syncthreads();
  #pragma unroll
  for (int j = 0; j < 4; ++j) {
    int n = ty * 4 + j;
    O[(long)(n0 + n) * K + k0 + tx] = f32_to_bf16_rne(tile[tx][n]);
  }
}

// ---------------- TDM: DMA a 128x32 bf16 tile into padded LDS [128][40] ----------------
#ifdef USE_TDM
__device__ __forceinline__ void tdm_load_tile_128x32(
    unsigned lds_off, const unsigned short* gptr, int row_stride_elems, int rows_avail)
{
  unsigned long long ga = (unsigned long long)(size_t)gptr;
  v4u g0;
  g0[0] = 1u;                                        // count=1, user descriptor, no gather
  g0[1] = lds_off;                                   // lds_addr (bytes)
  g0[2] = (unsigned)(ga & 0xFFFFFFFFu);              // global_addr[31:0]
  g0[3] = (unsigned)((ga >> 32) & 0x01FFFFFFu) | (2u << 30);  // global_addr[56:32], type=2
  v8i g1;
  g1[0] = (int)((1u << 16)      // data_size = 2 bytes
              | (1u << 20)      // pad_enable
              | (3u << 22)      // pad_interval: 16 DWORDs (one 32-elem row)
              | (3u << 25));    // pad_amount: 4 DWORDs (8 bf16 of padding)
  unsigned td0 = (unsigned)row_stride_elems;         // tensor_dim0 (elements)
  unsigned td1 = (unsigned)rows_avail;               // tensor_dim1 (rows)
  g1[1] = (int)((td0 & 0xFFFFu) << 16);              // atomic_addr=0 | tensor_dim0[15:0]
  g1[2] = (int)((td0 >> 16) | ((td1 & 0xFFFFu) << 16));
  g1[3] = (int)((td1 >> 16) | (32u << 16));          // tensor_dim1[31:16] | tile_dim0=32
  g1[4] = (int)128u;                                  // tile_dim1=128, tile_dim2=0
  unsigned long long st0 = (unsigned long long)row_stride_elems;
  g1[5] = (int)(st0 & 0xFFFFFFFFu);                  // tensor_dim0_stride[31:0]
  g1[6] = (int)((st0 >> 32) & 0xFFFFu);              // dim0_stride[47:32] | dim1_stride[15:0]=0
  g1[7] = 0;
  v4i z4 = {0, 0, 0, 0};
  v8i z8 = {0, 0, 0, 0, 0, 0, 0, 0};
  __builtin_amdgcn_tensor_load_to_lds(g0, g1, z4, z4, z8, 0);
}
#endif

// ---------------- bf16 WMMA GEMM ----------------
// Per batch b: C = A[b] (MxK, bf16 row-major) x W[e(b)]^T  (Wt is NxK row-major)
// Block tile: 128 (M) x 128 (N); 8 waves, each wave: 16 x 128 (8 accumulators).
// mode 0: relu -> bf16 output;  mode 1: raw f32 output
__global__ __launch_bounds__(256) void gemm_bf16_wmma(
    const unsigned short* __restrict__ A, long a_bstride,
    const unsigned short* __restrict__ Wt, long w_estride,
    const int* __restrict__ labels, int use_label, int fixed_e,
    int M, int N, int K,
    void* __restrict__ Out, long o_bstride, int mode)
{
  __shared__ unsigned short As[128][40];  // 128 x 32 (+8 pad), rows = M
  __shared__ unsigned short Bs[128][40];  // 128 x 32 (+8 pad), rows = N (Wt is N x K)

  const int b    = blockIdx.z;
  const int m0   = blockIdx.y * 128;
  const int n0   = blockIdx.x * 128;
  const int tid  = threadIdx.x;
  const int wave = tid >> 5;
  const int lane = tid & 31;

  const unsigned short* Ab = A + (long)b * a_bstride;
  const int e = use_label ? labels[b] : fixed_e;
  const unsigned short* Wb = Wt + (long)e * w_estride;

  v8f acc[8] = {};

  for (int k0 = 0; k0 < K; k0 += 32) {
#ifdef USE_TDM
    // ---- TDM staging: wave 0 DMAs the A tile, wave 1 DMAs the B tile ----
    if (wave == 0) {
      tdm_load_tile_128x32((unsigned)(unsigned long long)(size_t)&As[0][0],
                           Ab + (long)m0 * K + k0, K, M - m0);
      __builtin_amdgcn_s_wait_tensorcnt((short)0);
    } else if (wave == 1) {
      tdm_load_tile_128x32((unsigned)(unsigned long long)(size_t)&Bs[0][0],
                           Wb + (long)n0 * K + k0, K, N - n0);
      __builtin_amdgcn_s_wait_tensorcnt((short)0);
    }
    __syncthreads();
#else
    // ---- manual staging: both tiles are contiguous-K, pure b128 copies ----
    #pragma unroll
    for (int r = 0; r < 2; ++r) {
      int i = tid + r * 256;
      int row = i >> 2, seg = i & 3;
      *(uint4*)&As[row][seg * 8] =
          *(const uint4*)(Ab + (long)(m0 + row) * K + k0 + seg * 8);
      *(uint4*)&Bs[row][seg * 8] =
          *(const uint4*)(Wb + (long)(n0 + row) * K + k0 + seg * 8);
    }
    __syncthreads();
    if (k0 + 32 < K) {
      __builtin_prefetch(Ab + (long)(m0 + (tid >> 1)) * K + (k0 + 32) + (tid & 1) * 16, 0, 1);
      __builtin_prefetch(Wb + (long)(n0 + (tid >> 1)) * K + (k0 + 32) + (tid & 1) * 16, 0, 1);
    }
#endif

    // ---- fragments + WMMA ----
    // A 16x32 bf16: lanes 0-15 -> M=lane, K=0..7 & 16..23; lanes 16-31 -> K=8..15 & 24..31
    union Frag { uint4 q[2]; v16bf v; } fa, fb;
    const int mloc = wave * 16 + (lane & 15);
    const int kb   = (lane >> 4) * 8;
    fa.q[0] = *(const uint4*)&As[mloc][kb];
    fa.q[1] = *(const uint4*)&As[mloc][kb + 16];
    // B 32x16: lanes 0-15 -> K=0..7 & 16..23, lanes 16-31 -> K=8..15 & 24..31, col = lane&15
    const int kb2 = (lane >> 4) * 8;
    const int ncl = lane & 15;
    #pragma unroll
    for (int a2 = 0; a2 < 8; ++a2) {
      fb.q[0] = *(const uint4*)&Bs[a2 * 16 + ncl][kb2];
      fb.q[1] = *(const uint4*)&Bs[a2 * 16 + ncl][kb2 + 16];
      acc[a2] = __builtin_amdgcn_wmma_f32_16x16x32_bf16(
          false, fa.v, false, fb.v, (short)0, acc[a2], false, false);
    }
    __syncthreads();
  }

  // C/D layout: VGPR r -> M = r (lanes 0-15) or r+8 (lanes 16-31), N = lane&15
  const int rowoff = (lane >> 4) * 8;
  const int col    = lane & 15;
  if (mode == 0) {
    unsigned short* O = (unsigned short*)Out + (long)b * o_bstride;
    #pragma unroll
    for (int a2 = 0; a2 < 8; ++a2) {
      #pragma unroll
      for (int r = 0; r < 8; ++r) {
        int m = m0 + wave * 16 + rowoff + r;
        int n = n0 + a2 * 16 + col;
        float v = acc[a2][r];
        v = v > 0.0f ? v : 0.0f;                 // fused relu
        O[(long)m * N + n] = f32_to_bf16_rne(v);
      }
    }
  } else {
    float* O = (float*)Out + (long)b * o_bstride;
    #pragma unroll
    for (int a2 = 0; a2 < 8; ++a2) {
      #pragma unroll
      for (int r = 0; r < 8; ++r) {
        int m = m0 + wave * 16 + rowoff + r;
        int n = n0 + a2 * 16 + col;
        O[(long)m * N + n] = acc[a2][r];
      }
    }
  }
}

// ---------------- per-batch gate: softmax(relu(gate_in @ g1) @ g2) ----------------
__global__ __launch_bounds__(256) void gate_kernel(
    const float* __restrict__ eo, const float* __restrict__ s0, const float* __restrict__ s1,
    const float* __restrict__ g1, const float* __restrict__ g2,
    float* __restrict__ wout)
{
  __shared__ float gi[3 * D_];
  __shared__ float rr[D_];
  __shared__ float logits[3];
  const int b = blockIdx.x;
  const int tid = threadIdx.x;
  const long base = (long)b * S_ * D_;   // first token of batch b
  for (int i = tid; i < D_; i += 256) {
    gi[i]          = eo[base + i];
    gi[D_ + i]     = s0[base + i];
    gi[2 * D_ + i] = s1[base + i];
  }
  if (tid < 3) logits[tid] = 0.0f;
  __syncthreads();
  for (int j = tid; j < D_; j += 256) {
    float acc = 0.0f;
    for (int i = 0; i < 3 * D_; ++i) acc += gi[i] * g1[(long)i * D_ + j];
    rr[j] = acc > 0.0f ? acc : 0.0f;
  }
  __syncthreads();
  float p0 = 0.f, p1 = 0.f, p2 = 0.f;
  for (int j = tid; j < D_; j += 256) {
    float r = rr[j];
    p0 += r * g2[j * 3 + 0];
    p1 += r * g2[j * 3 + 1];
    p2 += r * g2[j * 3 + 2];
  }
  atomicAdd(&logits[0], p0);
  atomicAdd(&logits[1], p1);
  atomicAdd(&logits[2], p2);
  __syncthreads();
  if (tid == 0) {
    float l0 = logits[0], l1 = logits[1], l2 = logits[2];
    float mx = fmaxf(l0, fmaxf(l1, l2));
    float e0 = __expf(l0 - mx), e1 = __expf(l1 - mx), e2 = __expf(l2 - mx);
    float s = e0 + e1 + e2;
    wout[b * 3 + 0] = e0 / s;
    wout[b * 3 + 1] = e1 / s;
    wout[b * 3 + 2] = e2 / s;
  }
}

// ---------------- weighted combine ----------------
__global__ __launch_bounds__(256) void combine_kernel(
    const float* __restrict__ eo, const float* __restrict__ s0, const float* __restrict__ s1,
    const float* __restrict__ w, float* __restrict__ out)
{
  const long sd = (long)S_ * D_;
  long i = ((long)blockIdx.x * 256 + threadIdx.x) * 4;
  long b = i / sd;
  float w0 = w[b * 3 + 0], w1 = w[b * 3 + 1], w2 = w[b * 3 + 2];
  float4 a = *(const float4*)(eo + i);
  float4 c = *(const float4*)(s0 + i);
  float4 d = *(const float4*)(s1 + i);
  float4 o;
  o.x = a.x * w0 + c.x * w1 + d.x * w2;
  o.y = a.y * w0 + c.y * w1 + d.y * w2;
  o.z = a.z * w0 + c.z * w1 + d.z * w2;
  o.w = a.w * w0 + c.w * w1 + d.w * w2;
  *(float4*)(out + i) = o;
}

extern "C" void kernel_launch(void* const* d_in, const int* in_sizes, int n_in,
                              void* d_out, int out_size, void* d_ws, size_t ws_size,
                              hipStream_t stream) {
  const float* x   = (const float*)d_in[0];
  const int*   lbl = (const int*)  d_in[1];
  const float* wi  = (const float*)d_in[2];
  const float* wo  = (const float*)d_in[3];
  const float* swi = (const float*)d_in[4];
  const float* swo = (const float*)d_in[5];
  const float* g1  = (const float*)d_in[6];
  const float* g2  = (const float*)d_in[7];
  float* out = (float*)d_out;

  char* ws = (char*)d_ws;
  size_t off = 0;
  auto take = [&](size_t bytes) -> char* {
    char* p = ws + off;
    off += (bytes + 255) & ~(size_t)255;
    return p;
  };
  unsigned short* xbf    = (unsigned short*)take((size_t)B_ * S_ * D_ * 2);
  unsigned short* wibfT  = (unsigned short*)take((size_t)E_ * D_ * F_ * 2);  // (F x D) per expert
  unsigned short* wobfT  = (unsigned short*)take((size_t)E_ * F_ * D_ * 2);  // (D x F) per expert
  unsigned short* swibfT = (unsigned short*)take((size_t)2 * D_ * F_ * 2);
  unsigned short* swobfT = (unsigned short*)take((size_t)2 * F_ * D_ * 2);
  unsigned short* hbf    = (unsigned short*)take((size_t)B_ * S_ * F_ * 2);
  float* eo = (float*)take((size_t)B_ * S_ * D_ * 4);
  float* s0 = (float*)take((size_t)B_ * S_ * D_ * 4);
  float* s1 = (float*)take((size_t)B_ * S_ * D_ * 4);
  float* wg = (float*)take((size_t)B_ * 3 * 4);

  // conversions (weights transposed so GEMM B tiles are contiguous-K)
  cvt_f32_bf16<<<2048, 256, 0, stream>>>(x, xbf, (long)B_ * S_ * D_);
  cvt_transpose_f32_bf16<<<dim3(D_ / 32, F_ / 32, E_), 256, 0, stream>>>(wi,  wibfT,  D_, F_);
  cvt_transpose_f32_bf16<<<dim3(F_ / 32, D_ / 32, E_), 256, 0, stream>>>(wo,  wobfT,  F_, D_);
  cvt_transpose_f32_bf16<<<dim3(D_ / 32, F_ / 32, 2 ), 256, 0, stream>>>(swi, swibfT, D_, F_);
  cvt_transpose_f32_bf16<<<dim3(F_ / 32, D_ / 32, 2 ), 256, 0, stream>>>(swo, swobfT, F_, D_);

  dim3 blk(256);
  dim3 grid1(F_ / 128, S_ / 128, B_);  // GEMM1: x(SxD) @ Wi(DxF)
  dim3 grid2(D_ / 128, S_ / 128, B_);  // GEMM2: h(SxF) @ Wo(FxD)

  // expert path (per-batch weight selection via type_label)
  gemm_bf16_wmma<<<grid1, blk, 0, stream>>>(xbf, (long)S_ * D_, wibfT, (long)D_ * F_,
                                            lbl, 1, 0, S_, F_, D_, hbf, (long)S_ * F_, 0);
  gemm_bf16_wmma<<<grid2, blk, 0, stream>>>(hbf, (long)S_ * F_, wobfT, (long)F_ * D_,
                                            lbl, 1, 0, S_, D_, F_, eo, (long)S_ * D_, 1);
  // shared path 0
  gemm_bf16_wmma<<<grid1, blk, 0, stream>>>(xbf, (long)S_ * D_, swibfT, (long)D_ * F_,
                                            lbl, 0, 0, S_, F_, D_, hbf, (long)S_ * F_, 0);
  gemm_bf16_wmma<<<grid2, blk, 0, stream>>>(hbf, (long)S_ * F_, swobfT, (long)F_ * D_,
                                            lbl, 0, 0, S_, D_, F_, s0, (long)S_ * D_, 1);
  // shared path 1
  gemm_bf16_wmma<<<grid1, blk, 0, stream>>>(xbf, (long)S_ * D_, swibfT, (long)D_ * F_,
                                            lbl, 0, 1, S_, F_, D_, hbf, (long)S_ * F_, 0);
  gemm_bf16_wmma<<<grid2, blk, 0, stream>>>(hbf, (long)S_ * F_, swobfT, (long)F_ * D_,
                                            lbl, 0, 1, S_, D_, F_, s1, (long)S_ * D_, 1);

  gate_kernel<<<B_, 256, 0, stream>>>(eo, s0, s1, g1, g2, wg);

  unsigned nblk = (unsigned)(((size_t)B_ * S_ * D_) / 1024);
  combine_kernel<<<nblk, 256, 0, stream>>>(eo, s0, s1, wg, out);
}
